// IDXST_1133871366820
// MI455X (gfx1250) — compile-verified, matrix operand order checked by reading, not run on hardware
//
#include <hip/hip_runtime.h>

// IDXST as GEMM: Y[r,k] = sum_n X[r,n] * sin(pi * n*(2k+1) / (2N)), N = 4096.
// bf16x3 split GEMM on the CDNA5 WMMA pipe (v_wmma_f32_16x16x32_bf16).
// S (sine matrix) and X are decomposed ONCE into bf16 hi/lo operand-ready
// tiles in workspace. The GEMM hot loop double-buffers those tiles into LDS
// with GLOBAL_LOAD_ASYNC_TO_LDS_B128 (ASYNCcnt) — no VGPR staging — and runs
// 24 WMMAs per wave per k-step.

typedef __attribute__((ext_vector_type(16))) __bf16 v16bf;
typedef __attribute__((ext_vector_type(8)))  float  v8f;

#define NDIM   4096
#define KSTEPS (NDIM / 32)       // 128 k-steps of 32
#define LSTRH  24                // halfs per lane-row in LDS (16 used + 8 pad = 48B)

union V16 {
    uint4 u[2];
    v16bf v;
};
union Pack8 {
    uint4  q;
    __bf16 h[8];
};

// WMMA 16x16x32 bf16 per-lane operand layout (A: row m, B: col n symmetric):
//   lane = (mn & 15) | (((k>>3)&1) << 4)
//   slot = (k & 7)   | (((k>>4)&1) << 3)
__device__ __forceinline__ int lds_lane(int mn, int k) {
    return (mn & 15) | (((k >> 3) & 1) << 4);
}
__device__ __forceinline__ int lds_slot(int k) {
    return (k & 7) | (((k >> 4) & 1) << 3);
}

// Low 32 bits of a generic pointer to LDS == LDS byte offset (aperture in high bits).
__device__ __forceinline__ unsigned lds_off(const void* p) {
    return (unsigned)(uintptr_t)p;
}

__device__ __forceinline__ void async_b128(unsigned ldsAddr, const void* gaddr) {
    asm volatile("global_load_async_to_lds_b128 %0, %1, off"
                 :: "v"(ldsAddr), "v"(gaddr) : "memory");
}

// ---------------------------------------------------------------------------
// Precompute kernels: operand-ready global tiles.
// Chunk index gidx (2^21): [tile(8b) | kstep(7b) | lane(5b) | c2(1b)];
// chunk = 8 consecutive halfs of one lane's 16-half operand row, covering
// k = c2*16 + (lane>>4)*8 + (0..7)  (contiguous in k).
// ---------------------------------------------------------------------------
__global__ __launch_bounds__(256)
void gen_S_kernel(__bf16* __restrict__ Sh, __bf16* __restrict__ Sl) {
    const unsigned gidx = blockIdx.x * 256u + threadIdx.x;
    const int c2    = gidx & 1;
    const int lane  = (gidx >> 1) & 31;
    const int kstep = (gidx >> 6) & 127;
    const int ct    = gidx >> 13;               // 0..255
    const int c     = ct * 16 + (lane & 15);    // output column index
    const int kg0   = kstep * 32 + c2 * 16 + (lane >> 4) * 8;
    const unsigned wmul = (unsigned)(2 * c + 1);

    Pack8 ph, pl;
#pragma unroll
    for (int e = 0; e < 8; ++e) {
        unsigned t = ((unsigned)(kg0 + e) * wmul) & 16383u;   // exact phase mod 4N
        float s = sinpif((float)t * (1.0f / 8192.0f));
        __bf16 hi = (__bf16)s;
        ph.h[e] = hi;
        pl.h[e] = (__bf16)(s - (float)hi);
    }
    ((uint4*)Sh)[gidx] = ph.q;
    ((uint4*)Sl)[gidx] = pl.q;
}

__global__ __launch_bounds__(256)
void gen_A_kernel(const float* __restrict__ X,
                  __bf16* __restrict__ Ah, __bf16* __restrict__ Al) {
    const unsigned gidx = blockIdx.x * 256u + threadIdx.x;
    const int c2    = gidx & 1;
    const int lane  = (gidx >> 1) & 31;
    const int kstep = (gidx >> 6) & 127;
    const int rt    = gidx >> 13;               // 0..255
    const int row   = rt * 16 + (lane & 15);
    const int kg0   = kstep * 32 + c2 * 16 + (lane >> 4) * 8;

    const float4 v0 = *(const float4*)(X + (size_t)row * NDIM + kg0);
    const float4 v1 = *(const float4*)(X + (size_t)row * NDIM + kg0 + 4);
    float xv[8] = {v0.x, v0.y, v0.z, v0.w, v1.x, v1.y, v1.z, v1.w};

    Pack8 ph, pl;
#pragma unroll
    for (int e = 0; e < 8; ++e) {
        __bf16 hi = (__bf16)xv[e];
        ph.h[e] = hi;
        pl.h[e] = (__bf16)(xv[e] - (float)hi);
    }
    ((uint4*)Ah)[gidx] = ph.q;
    ((uint4*)Al)[gidx] = pl.q;
}

// ---------------------------------------------------------------------------
// Fast GEMM: block tile 256(M) x 128(N), BK = 32, 512 threads = 16 waves
// (8 M x 2 N); each wave owns 32x64 = 2x4 WMMA tiles x 3 bf16 passes.
// Tiles stream in via async-to-LDS with 2-deep double buffering.
// ---------------------------------------------------------------------------
__global__ __launch_bounds__(512)
void gemm_async_kernel(const __bf16* __restrict__ Ah, const __bf16* __restrict__ Al,
                       const __bf16* __restrict__ Sh, const __bf16* __restrict__ Sl,
                       float* __restrict__ Y) {
    // [buf][hi/lo][tile][lane][chunk]; chunk 2 of 3 is 16B bank padding
    __shared__ uint4 sA[2][2][16][32][3];   // 2 x 48KB
    __shared__ uint4 sB[2][2][8][32][3];    // 2 x 24KB

    const int tid  = threadIdx.x;
    const int lane = tid & 31;
    const int w    = tid >> 5;   // 16 waves
    const int wm   = w >> 1;     // 0..7
    const int wn   = w & 1;      // 0..1

    const int rowBase = blockIdx.y * 256;
    const int colBase = blockIdx.x * 128;

    // Per-thread async copy descriptors (4 A-chunks + 2 B-chunks per k-step).
    const __bf16* agsrc[4];
    unsigned      alds0[4];
#pragma unroll
    for (int j = 0; j < 4; ++j) {
        int idx = j * 512 + tid;             // 2048 uint4 for A hi+lo
        int u   = idx & 1;
        int ll  = (idx >> 1) & 31;
        int rt  = (idx >> 6) & 15;
        int sel = idx >> 10;                 // 0: hi, 1: lo
        const __bf16* src = sel ? Al : Ah;
        agsrc[j] = src + ((size_t)(blockIdx.y * 16 + rt) * KSTEPS) * 512
                       + (size_t)ll * 16 + u * 8;
        alds0[j] = lds_off(&sA[0][sel][rt][ll][u]);
    }
    const __bf16* bgsrc[2];
    unsigned      blds0[2];
#pragma unroll
    for (int j = 0; j < 2; ++j) {
        int idx = j * 512 + tid;             // 1024 uint4 for B hi+lo
        int u   = idx & 1;
        int ll  = (idx >> 1) & 31;
        int ct  = (idx >> 6) & 7;
        int sel = idx >> 9;                  // 0: hi, 1: lo
        const __bf16* src = sel ? Sl : Sh;
        bgsrc[j] = src + ((size_t)(blockIdx.x * 8 + ct) * KSTEPS) * 512
                       + (size_t)ll * 16 + u * 8;
        blds0[j] = lds_off(&sB[0][sel][ct][ll][u]);
    }
    const unsigned aStride = (unsigned)sizeof(sA[0]);  // 49152
    const unsigned bStride = (unsigned)sizeof(sB[0]);  // 24576

    auto issue = [&](int buf, int it) {
#pragma unroll
        for (int j = 0; j < 4; ++j)
            async_b128(alds0[j] + (unsigned)buf * aStride, agsrc[j] + (size_t)it * 512);
#pragma unroll
        for (int j = 0; j < 2; ++j)
            async_b128(blds0[j] + (unsigned)buf * bStride, bgsrc[j] + (size_t)it * 512);
    };

    v8f acc[2][4];
    const v8f vzero = {0.f, 0.f, 0.f, 0.f, 0.f, 0.f, 0.f, 0.f};
#pragma unroll
    for (int rm = 0; rm < 2; ++rm)
#pragma unroll
        for (int cn = 0; cn < 4; ++cn)
            acc[rm][cn] = vzero;

    issue(0, 0);

    for (int it = 0; it < KSTEPS; ++it) {
        const int cur = it & 1;
        if (it + 1 < KSTEPS) {
            issue(cur ^ 1, it + 1);
            // 6 ops of group it+1 in flight; group it complete (in-order).
            asm volatile("s_wait_asynccnt 0x6" ::: "memory");
        } else {
            asm volatile("s_wait_asynccnt 0x0" ::: "memory");
        }
        __syncthreads();   // all waves' group-it writes visible

        V16 ahi[2], alo[2];
#pragma unroll
        for (int rm = 0; rm < 2; ++rm) {
            int rt = wm * 2 + rm;
            ahi[rm].u[0] = sA[cur][0][rt][lane][0];
            ahi[rm].u[1] = sA[cur][0][rt][lane][1];
            alo[rm].u[0] = sA[cur][1][rt][lane][0];
            alo[rm].u[1] = sA[cur][1][rt][lane][1];
        }
#pragma unroll
        for (int cn = 0; cn < 4; ++cn) {
            int ct = wn * 4 + cn;
            V16 bhi, blo;
            bhi.u[0] = sB[cur][0][ct][lane][0];
            bhi.u[1] = sB[cur][0][ct][lane][1];
            blo.u[0] = sB[cur][1][ct][lane][0];
            blo.u[1] = sB[cur][1][ct][lane][1];
#pragma unroll
            for (int rm = 0; rm < 2; ++rm) {
                acc[rm][cn] = __builtin_amdgcn_wmma_f32_16x16x32_bf16(
                    false, ahi[rm].v, false, bhi.v, (short)0, acc[rm][cn], false, false);
                acc[rm][cn] = __builtin_amdgcn_wmma_f32_16x16x32_bf16(
                    false, alo[rm].v, false, bhi.v, (short)0, acc[rm][cn], false, false);
                acc[rm][cn] = __builtin_amdgcn_wmma_f32_16x16x32_bf16(
                    false, ahi[rm].v, false, blo.v, (short)0, acc[rm][cn], false, false);
            }
        }
        __syncthreads();   // buffer 'cur' free for group it+2
    }

    // ---- store C: VGPR j of v8f holds row ((lane>>4)*8 + j), col (lane & 15)
    const int cRow = (lane >> 4) << 3;
    const int cCol = lane & 15;
#pragma unroll
    for (int rm = 0; rm < 2; ++rm) {
#pragma unroll
        for (int cn = 0; cn < 4; ++cn) {
            int r0 = rowBase + wm * 32 + rm * 16 + cRow;
            int c0 = colBase + wn * 64 + cn * 16 + cCol;
#pragma unroll
            for (int jj = 0; jj < 8; ++jj) {
                Y[(size_t)(r0 + jj) * NDIM + c0] = acc[rm][cn][jj];
            }
        }
    }
}

// ---------------------------------------------------------------------------
// Mid tier: S precomputed, A split from X on the fly. 256x128 tile, 512 thr.
// ---------------------------------------------------------------------------
__global__ __launch_bounds__(512)
void gemm_preB_kernel(const __bf16* __restrict__ Sh, const __bf16* __restrict__ Sl,
                      const float* __restrict__ X, float* __restrict__ Y) {
    __shared__ uint4 sAhi[16][32][3];
    __shared__ uint4 sAlo[16][32][3];
    __shared__ uint4 sBhi[8][32][3];
    __shared__ uint4 sBlo[8][32][3];

    const int tid  = threadIdx.x;
    const int lane = tid & 31;
    const int w    = tid >> 5;
    const int wm   = w >> 1;
    const int wn   = w & 1;

    const int rowBase = blockIdx.y * 256;
    const int colBase = blockIdx.x * 128;

    v8f acc[2][4];
    const v8f vzero = {0.f, 0.f, 0.f, 0.f, 0.f, 0.f, 0.f, 0.f};
#pragma unroll
    for (int rm = 0; rm < 2; ++rm)
#pragma unroll
        for (int cn = 0; cn < 4; ++cn)
            acc[rm][cn] = vzero;

    for (int it = 0; it < KSTEPS; ++it) {
        const int k0 = it * 32;
        __bf16* aHi = (__bf16*)sAhi;
        __bf16* aLo = (__bf16*)sAlo;
#pragma unroll
        for (int j = 0; j < 4; ++j) {
            int idx4 = j * 512 + tid;
            int row  = idx4 >> 3;
            int kq   = (idx4 & 7) << 2;
            const float4 vx =
                *(const float4*)(X + (size_t)(rowBase + row) * NDIM + (k0 + kq));
            float xv[4] = {vx.x, vx.y, vx.z, vx.w};
            int rt = row >> 4, m = row & 15;
#pragma unroll
            for (int e = 0; e < 4; ++e) {
                int k = kq + e;
                __bf16 hi = (__bf16)xv[e];
                __bf16 lo = (__bf16)(xv[e] - (float)hi);
                int off = (rt * 32 + lds_lane(m, k)) * LSTRH + lds_slot(k);
                aHi[off] = hi;
                aLo[off] = lo;
            }
        }
#pragma unroll
        for (int j = 0; j < 2; ++j) {
            int idx = j * 512 + tid;
            int u   = idx & 1;
            int ll  = (idx >> 1) & 31;
            int ct  = (idx >> 6) & 7;
            int sel = idx >> 9;
            const __bf16* src = sel ? Sl : Sh;
            size_t goff = (((size_t)(blockIdx.x * 8 + ct) * KSTEPS + it) * 512)
                          + (size_t)ll * 16 + u * 8;
            uint4 v = *(const uint4*)(src + goff);
            (sel ? sBlo : sBhi)[ct][ll][u] = v;
        }
        __syncthreads();

        V16 ahi[2], alo[2];
#pragma unroll
        for (int rm = 0; rm < 2; ++rm) {
            int rt = wm * 2 + rm;
            ahi[rm].u[0] = sAhi[rt][lane][0];
            ahi[rm].u[1] = sAhi[rt][lane][1];
            alo[rm].u[0] = sAlo[rt][lane][0];
            alo[rm].u[1] = sAlo[rt][lane][1];
        }
#pragma unroll
        for (int cn = 0; cn < 4; ++cn) {
            int ct = wn * 4 + cn;
            V16 bhi, blo;
            bhi.u[0] = sBhi[ct][lane][0];
            bhi.u[1] = sBhi[ct][lane][1];
            blo.u[0] = sBlo[ct][lane][0];
            blo.u[1] = sBlo[ct][lane][1];
#pragma unroll
            for (int rm = 0; rm < 2; ++rm) {
                acc[rm][cn] = __builtin_amdgcn_wmma_f32_16x16x32_bf16(
                    false, ahi[rm].v, false, bhi.v, (short)0, acc[rm][cn], false, false);
                acc[rm][cn] = __builtin_amdgcn_wmma_f32_16x16x32_bf16(
                    false, alo[rm].v, false, bhi.v, (short)0, acc[rm][cn], false, false);
                acc[rm][cn] = __builtin_amdgcn_wmma_f32_16x16x32_bf16(
                    false, ahi[rm].v, false, blo.v, (short)0, acc[rm][cn], false, false);
            }
        }
        __syncthreads();
    }

    const int cRow = (lane >> 4) << 3;
    const int cCol = lane & 15;
#pragma unroll
    for (int rm = 0; rm < 2; ++rm) {
#pragma unroll
        for (int cn = 0; cn < 4; ++cn) {
            int r0 = rowBase + wm * 32 + rm * 16 + cRow;
            int c0 = colBase + wn * 64 + cn * 16 + cCol;
#pragma unroll
            for (int jj = 0; jj < 8; ++jj) {
                Y[(size_t)(r0 + jj) * NDIM + c0] = acc[rm][cn][jj];
            }
        }
    }
}

// ---------------------------------------------------------------------------
// Fallback (tiny workspace): everything on the fly. 128x128 tile, 256 thr.
// ---------------------------------------------------------------------------
__global__ __launch_bounds__(256)
void idxst_fly_kernel(const float* __restrict__ X, float* __restrict__ Y) {
    __shared__ uint4 sAhi[8][32][3];
    __shared__ uint4 sAlo[8][32][3];
    __shared__ uint4 sBhi[8][32][3];
    __shared__ uint4 sBlo[8][32][3];

    __bf16* aHi = (__bf16*)sAhi;
    __bf16* aLo = (__bf16*)sAlo;
    __bf16* bHi = (__bf16*)sBhi;
    __bf16* bLo = (__bf16*)sBlo;

    const int tid  = threadIdx.x;
    const int lane = tid & 31;
    const int w    = tid >> 5;
    const int wm   = w >> 1;
    const int wn   = w & 1;

    const int rowBase = blockIdx.y * 128;
    const int colBase = blockIdx.x * 128;

    v8f acc[2][4];
    const v8f vzero = {0.f, 0.f, 0.f, 0.f, 0.f, 0.f, 0.f, 0.f};
#pragma unroll
    for (int rm = 0; rm < 2; ++rm)
#pragma unroll
        for (int cn = 0; cn < 4; ++cn)
            acc[rm][cn] = vzero;

    for (int it = 0; it < KSTEPS; ++it) {
        const int k0 = it * 32;
#pragma unroll
        for (int j = 0; j < 4; ++j) {
            int idx4 = j * 256 + tid;
            int row  = idx4 >> 3;
            int kq   = (idx4 & 7) << 2;
            const float4 vx =
                *(const float4*)(X + (size_t)(rowBase + row) * NDIM + (k0 + kq));
            float xv[4] = {vx.x, vx.y, vx.z, vx.w};
            int rt = row >> 4, m = row & 15;
#pragma unroll
            for (int e = 0; e < 4; ++e) {
                int k = kq + e;
                __bf16 hi = (__bf16)xv[e];
                __bf16 lo = (__bf16)(xv[e] - (float)hi);
                int off = (rt * 32 + lds_lane(m, k)) * LSTRH + lds_slot(k);
                aHi[off] = hi;
                aLo[off] = lo;
            }
        }
#pragma unroll
        for (int j = 0; j < 16; ++j) {
            int idx = j * 256 + tid;
            int k   = idx >> 7;
            int n   = idx & 127;
            unsigned t = ((unsigned)(k0 + k) * (unsigned)(2 * (colBase + n) + 1)) & 16383u;
            float s = sinpif((float)t * (1.0f / 8192.0f));
            __bf16 hi = (__bf16)s;
            __bf16 lo = (__bf16)(s - (float)hi);
            int off = ((n >> 4) * 32 + lds_lane(n, k)) * LSTRH + lds_slot(k);
            bHi[off] = hi;
            bLo[off] = lo;
        }
        __syncthreads();

        V16 ahi[2], alo[2];
#pragma unroll
        for (int rm = 0; rm < 2; ++rm) {
            int rt = wm * 2 + rm;
            ahi[rm].u[0] = sAhi[rt][lane][0];
            ahi[rm].u[1] = sAhi[rt][lane][1];
            alo[rm].u[0] = sAlo[rt][lane][0];
            alo[rm].u[1] = sAlo[rt][lane][1];
        }
#pragma unroll
        for (int cn = 0; cn < 4; ++cn) {
            int ct = wn * 4 + cn;
            V16 bhi, blo;
            bhi.u[0] = sBhi[ct][lane][0];
            bhi.u[1] = sBhi[ct][lane][1];
            blo.u[0] = sBlo[ct][lane][0];
            blo.u[1] = sBlo[ct][lane][1];
#pragma unroll
            for (int rm = 0; rm < 2; ++rm) {
                acc[rm][cn] = __builtin_amdgcn_wmma_f32_16x16x32_bf16(
                    false, ahi[rm].v, false, bhi.v, (short)0, acc[rm][cn], false, false);
                acc[rm][cn] = __builtin_amdgcn_wmma_f32_16x16x32_bf16(
                    false, alo[rm].v, false, bhi.v, (short)0, acc[rm][cn], false, false);
                acc[rm][cn] = __builtin_amdgcn_wmma_f32_16x16x32_bf16(
                    false, ahi[rm].v, false, blo.v, (short)0, acc[rm][cn], false, false);
            }
        }
        __syncthreads();
    }

    const int cRow = (lane >> 4) << 3;
    const int cCol = lane & 15;
#pragma unroll
    for (int rm = 0; rm < 2; ++rm) {
#pragma unroll
        for (int cn = 0; cn < 4; ++cn) {
            int r0 = rowBase + wm * 32 + rm * 16 + cRow;
            int c0 = colBase + wn * 64 + cn * 16 + cCol;
#pragma unroll
            for (int jj = 0; jj < 8; ++jj) {
                Y[(size_t)(r0 + jj) * NDIM + c0] = acc[rm][cn][jj];
            }
        }
    }
}

// ---------------------------------------------------------------------------
extern "C" void kernel_launch(void* const* d_in, const int* in_sizes, int n_in,
                              void* d_out, int out_size, void* d_ws, size_t ws_size,
                              hipStream_t stream) {
    (void)in_sizes; (void)n_in; (void)out_size;
    const float* X = (const float*)d_in[0];
    float*       Y = (float*)d_out;

    const size_t elems  = (size_t)NDIM * NDIM;          // 16.78M
    const size_t oneBuf = elems * sizeof(__bf16);       // 32 MB
    const int    genBlocks = (int)(elems / 8 / 256);    // 8192

    if (ws_size >= 4 * oneBuf) {
        // Full precompute: S hi/lo + A hi/lo (128 MB) -> async double-buffered GEMM
        __bf16* Sh = (__bf16*)d_ws;
        __bf16* Sl = Sh + elems;
        __bf16* Ah = Sl + elems;
        __bf16* Al = Ah + elems;
        gen_S_kernel<<<genBlocks, 256, 0, stream>>>(Sh, Sl);
        gen_A_kernel<<<genBlocks, 256, 0, stream>>>(X, Ah, Al);
        gemm_async_kernel<<<dim3(NDIM / 128, NDIM / 256), 512, 0, stream>>>(
            Ah, Al, Sh, Sl, Y);
    } else if (ws_size >= 2 * oneBuf) {
        // S precomputed only (64 MB); A split on the fly
        __bf16* Sh = (__bf16*)d_ws;
        __bf16* Sl = Sh + elems;
        gen_S_kernel<<<genBlocks, 256, 0, stream>>>(Sh, Sl);
        gemm_preB_kernel<<<dim3(NDIM / 128, NDIM / 256), 512, 0, stream>>>(
            Sh, Sl, X, Y);
    } else {
        // Tiny workspace: everything on the fly
        idxst_fly_kernel<<<dim3(NDIM / 128, NDIM / 128), 256, 0, stream>>>(X, Y);
    }
}
